// LSTM_19748259627488
// MI455X (gfx1250) — matrix operand order repaired and checked by reading, not compiled, not run on hardware
//
#include <hip/hip_runtime.h>
#include <hip/hip_bf16.h>

// Problem constants (from the reference)
#define BATCH 64
#define SEQ   512
#define EMB   300
#define EPAD  320      // E padded to a multiple of 32 for WMMA K-steps
#define HID   512
#define G4    2048     // 4*HID
#define NCLS  5
#define MTOT  (BATCH*SEQ)   // 32768 rows of the input-projection GEMM
#define RWG   32            // workgroups in the persistent recurrent kernel
#define XGBLK 4096          // floats per (t, wg) fragment block: 4 gates*4 mt*32 lanes*8

typedef __attribute__((ext_vector_type(16))) __bf16 v16bf;
typedef __attribute__((ext_vector_type(8)))  float  v8f;
typedef __attribute__((ext_vector_type(4)))  unsigned u32x4;
typedef __attribute__((ext_vector_type(8)))  unsigned u32x8;
typedef __hip_bfloat16 bf16_t;

union V16 { v16bf v; uint4 q[2]; };

// A-matrix fragment (16x32 bf16, M = lane%16):
//   lanes 0-15 hold K = [k0..k0+7] and [k0+16..k0+23]
//   lanes 16-31 hold K = [k0+8..k0+15] and [k0+24..k0+31]
__device__ inline v16bf load_fragA(const bf16_t* row, int k0, int lane) {
  int off = k0 + ((lane & 16) ? 8 : 0);
  V16 f;
  f.q[0] = *(const uint4*)(row + off);
  f.q[1] = *(const uint4*)(row + off + 16);
  return f.v;
}

// B-matrix fragment (32x16 bf16, N = lane%16): same K order as A fragment:
// lanes 0-15 K=[k0..k0+7],[k0+16..k0+23]; lanes 16-31 K=[k0+8..k0+15],[k0+24..k0+31].
// colK points at that column's K-major data.
__device__ inline v16bf load_fragB(const bf16_t* colK, int k0, int lane) {
  int off = k0 + ((lane & 16) ? 8 : 0);
  V16 f;
  f.q[0] = *(const uint4*)(colK + off);
  f.q[1] = *(const uint4*)(colK + off + 16);
  return f.v;
}

__device__ inline float sigm_f(float x)  { return 1.f / (1.f + __expf(-x)); }
__device__ inline float tanh_f(float x)  { return 2.f / (1.f + __expf(-2.f * x)) - 1.f; }

// ---------------------------------------------------------------------------
// TDM: 1-D async copy of `nelem` f32 from global to LDS (Tensor Data Mover).
// D# group0: count=1, lds_addr, global_addr[56:0], type=2 ("image").
// D# group1: wg_mask=0, data_size=2 (4B), tensor_dim0=tile_dim0=nelem,
//            tensor_dim1=1, tensor_dim0_stride=nelem.
// Inline asm keeps this portable across the 5-arg/6-arg builtin toolchains.
// Tracked with TENSORcnt.
// ---------------------------------------------------------------------------
__device__ inline void tdm_load_f32_1d(unsigned lds_byte_off, const float* src,
                                       unsigned nelem) {
  unsigned long long ga = (unsigned long long)(size_t)src;
  u32x4 g0;
  g0[0] = 1u;                                            // count=1, user-mode
  g0[1] = lds_byte_off;                                  // lds_addr (bytes)
  g0[2] = (unsigned)(ga & 0xffffffffull);                // global_addr[31:0]
  g0[3] = (unsigned)((ga >> 32) & 0x1ffffffull) | (2u << 30); // [56:32] | type=2
  u32x8 g1;
  g1[0] = (2u << 16);                                    // data_size = 4 bytes
  g1[1] = (nelem & 0xffffu) << 16;                       // tensor_dim0[15:0] @63:48
  g1[2] = ((nelem >> 16) & 0xffffu) | (1u << 16);        // dim0[31:16], tensor_dim1=1
  g1[3] = (nelem & 0xffffu) << 16;                       // tile_dim0 @127:112
  g1[4] = 0u;                                            // tile_dim1=0, tile_dim2=0
  g1[5] = nelem;                                         // tensor_dim0_stride[31:0]
  g1[6] = 0u;
  g1[7] = 0u;
  asm volatile("tensor_load_to_lds %0, %1" :: "s"(g0), "s"(g1) : "memory");
}

// ---------------------------------------------------------------------------
// Prep: f32 -> bf16 weight conversion (K-padded), bias fusion, h init,
// grid-barrier counter reset. Runs every call (deterministic).
// ---------------------------------------------------------------------------
__global__ void prep_kernel(const float* __restrict__ W_ih, const float* __restrict__ W_hh,
                            const float* __restrict__ b_ih, const float* __restrict__ b_hh,
                            bf16_t* __restrict__ wih16, bf16_t* __restrict__ whh16,
                            float* __restrict__ biasSum, bf16_t* __restrict__ hbuf,
                            unsigned* __restrict__ bar) {
  long stride = (long)gridDim.x * blockDim.x;
  long t0 = (long)blockIdx.x * blockDim.x + threadIdx.x;
  for (long i = t0; i < (long)G4 * EPAD; i += stride) {
    int g = (int)(i / EPAD), e = (int)(i % EPAD);
    wih16[i] = (e < EMB) ? __float2bfloat16(W_ih[(long)g * EMB + e]) : __float2bfloat16(0.f);
  }
  for (long i = t0; i < (long)G4 * HID; i += stride)
    whh16[i] = __float2bfloat16(W_hh[i]);
  for (long i = t0; i < G4; i += stride)
    biasSum[i] = b_ih[i] + b_hh[i];
  for (long i = t0; i < 2L * BATCH * HID; i += stride)   // h ping-pong buffers = 0
    hbuf[i] = __float2bfloat16(0.f);
  for (long i = t0; i < 64; i += stride) bar[i] = 0u;    // barrier count + generation
}

// ---------------------------------------------------------------------------
// Embedding gather + bf16 convert, TIME-MAJOR: emb16[s*64 + b][0..319]
// (time-major makes xproj C-fragment rows == batches of one timestep).
// ---------------------------------------------------------------------------
__global__ void gather_kernel(const int* __restrict__ x, const float* __restrict__ embw,
                              bf16_t* __restrict__ emb16) {
  long stride = (long)gridDim.x * blockDim.x;
  long t0 = (long)blockIdx.x * blockDim.x + threadIdx.x;
  for (long i = t0; i < (long)MTOT * EPAD; i += stride) {
    long row = i / EPAD; int e = (int)(i % EPAD);
    int s = (int)(row >> 6), b = (int)(row & 63);
    int tok = x[(long)b * SEQ + s];
    emb16[i] = (e < EMB) ? __float2bfloat16(embw[(long)tok * EMB + e]) : __float2bfloat16(0.f);
  }
}

// ---------------------------------------------------------------------------
// Input projection GEMM with fragment-order output:
//   xg[t][w][gate][mt][lane][r] = emb16[s*64+b] . wih16[gcol] + bias[gcol]
// Tile: 128(M) x 64(N) per 256-thread WG; weight rows staged once in LDS.
// Stores are 2x b128 per subtile (contiguous per (t,w) 16KB block).
// ---------------------------------------------------------------------------
__global__ __launch_bounds__(256)
void xproj_kernel(const bf16_t* __restrict__ emb16, const bf16_t* __restrict__ wih16,
                  const float* __restrict__ biasSum, float* __restrict__ xg) {
  __shared__ __align__(16) bf16_t wlds[64 * EPAD];          // 40 KB
  const int tileN0 = blockIdx.y * 64;

  const uint4* src = (const uint4*)(wih16 + (long)tileN0 * EPAD);
  uint4* dst = (uint4*)wlds;
  for (int i = threadIdx.x; i < 64 * EPAD / 8; i += blockDim.x) dst[i] = src[i];
  __syncthreads();

  const int wave = threadIdx.x >> 5, lane = threadIdx.x & 31;
  const int mbase = blockIdx.x * 128 + wave * 16;    // m = s*64 + b
  const bf16_t* arow = emb16 + (long)(mbase + (lane & 15)) * EPAD;
  __builtin_prefetch(arow, 0, 3);                    // global_prefetch_b8

  v8f acc[4];
  #pragma unroll
  for (int t = 0; t < 4; ++t)
    #pragma unroll
    for (int r = 0; r < 8; ++r) acc[t][r] = 0.f;

  for (int kk = 0; kk < EPAD; kk += 32) {
    v16bf a = load_fragA(arow, kk, lane);
    #pragma unroll
    for (int t = 0; t < 4; ++t) {
      v16bf b = load_fragB(wlds + (t * 16 + (lane & 15)) * EPAD, kk, lane);
      acc[t] = __builtin_amdgcn_wmma_f32_16x16x32_bf16(
          false, a, false, b, (short)0, acc[t], false, false);
    }
  }

  const int s_idx = mbase >> 6;          // timestep of this wave's 16 rows
  const int mt    = (mbase & 63) >> 4;   // batch tile 0..3
  #pragma unroll
  for (int tsub = 0; tsub < 4; ++tsub) {
    int gcol = tileN0 + tsub * 16 + (lane & 15);
    int g = gcol >> 9;                   // gate 0..3
    int w = (gcol & 511) >> 4;           // hid-block (recurrent WG) 0..31
    float bv = biasSum[gcol];
    float4 lo, hi;
    lo.x = acc[tsub][0] + bv; lo.y = acc[tsub][1] + bv;
    lo.z = acc[tsub][2] + bv; lo.w = acc[tsub][3] + bv;
    hi.x = acc[tsub][4] + bv; hi.y = acc[tsub][5] + bv;
    hi.z = acc[tsub][6] + bv; hi.w = acc[tsub][7] + bv;
    float4* dstp = (float4*)(xg + ((((long)s_idx * RWG + w) * 4 + g) * 4 + mt) * 256
                                + lane * 8);
    dstp[0] = lo; dstp[1] = hi;
  }
}

// ---------------------------------------------------------------------------
// Persistent LSTM recurrence. 32 WGs x 128 threads (4 waves). WG b owns hidden
// units [16b,16b+16): W_hh columns for all 4 gates in LDS (64 KB). Each step's
// 16 KB xg fragment block is TDM-prefetched into an LDS double buffer one step
// ahead (wave 0 issues tensor_load_to_lds; s_wait_tensorcnt 1 ensures the
// previous transfer landed while the next is in flight). Gates, c and the
// time-max pool live entirely in registers; h ping-pongs through global with
// one software grid barrier per timestep.
// ---------------------------------------------------------------------------
__global__ __launch_bounds__(128)
void lstm_recurrent(const bf16_t* __restrict__ whh16, const float* __restrict__ xg,
                    bf16_t* __restrict__ hbuf, float* __restrict__ hpool,
                    unsigned* __restrict__ bar_cnt, unsigned* __restrict__ bar_gen) {
  __shared__ __align__(16) bf16_t wlds[64 * HID];          // 64 KB
  __shared__ __align__(16) float  xglds[2][XGBLK];         // 2 x 16 KB double buffer
  const int hid0 = blockIdx.x * 16;

  for (int i = threadIdx.x; i < 64 * (HID / 8); i += blockDim.x) {
    int col = i / (HID / 8), chunk = i % (HID / 8);
    int g = col >> 4, n = col & 15;
    ((uint4*)wlds)[col * (HID / 8) + chunk] =
        ((const uint4*)(whh16 + (long)(g * HID + hid0 + n) * HID))[chunk];
  }

  const int wave = threadIdx.x >> 5, lane = threadIdx.x & 31;
  const int mbase = wave * 16;
  const int hidc = hid0 + (lane & 15);

  // Prologue: async-load step 0's xg block.
  if (wave == 0) {
    tdm_load_f32_1d((unsigned)(size_t)&xglds[0][0],
                    xg + ((long)0 * RWG + blockIdx.x) * XGBLK, XGBLK);
  }
  __syncthreads();

  v8f c, pool;
  #pragma unroll
  for (int r = 0; r < 8; ++r) { c[r] = 0.f; pool[r] = -3.0e38f; }

  for (int t = 0; t < SEQ; ++t) {
    const bf16_t* hread = hbuf + (long)(t & 1) * BATCH * HID;
    bf16_t*       hwrite = hbuf + (long)((t + 1) & 1) * BATCH * HID;

    // Kick off next step's xg DMA, then guarantee this step's block is in LDS.
    if (wave == 0) {
      if (t + 1 < SEQ) {
        tdm_load_f32_1d((unsigned)(size_t)&xglds[(t + 1) & 1][0],
                        xg + ((long)(t + 1) * RWG + blockIdx.x) * XGBLK, XGBLK);
        __builtin_amdgcn_s_wait_tensorcnt(1);
      } else {
        __builtin_amdgcn_s_wait_tensorcnt(0);
      }
    }
    __syncthreads();

    // Seed accumulators from the LDS fragment block (2x b128 per gate).
    v8f acc[4];
    #pragma unroll
    for (int g = 0; g < 4; ++g) {
      const float4* sp = (const float4*)&xglds[t & 1][((g * 4 + wave) * 32 + lane) * 8];
      float4 lo = sp[0], hi = sp[1];
      acc[g][0] = lo.x; acc[g][1] = lo.y; acc[g][2] = lo.z; acc[g][3] = lo.w;
      acc[g][4] = hi.x; acc[g][5] = hi.y; acc[g][6] = hi.z; acc[g][7] = hi.w;
    }

    const bf16_t* arow = hread + (long)(mbase + (lane & 15)) * HID;
    for (int kk = 0; kk < HID; kk += 32) {
      v16bf a = load_fragA(arow, kk, lane);
      #pragma unroll
      for (int g = 0; g < 4; ++g) {
        v16bf b = load_fragB(wlds + (g * 16 + (lane & 15)) * HID, kk, lane);
        acc[g] = __builtin_amdgcn_wmma_f32_16x16x32_bf16(
            false, a, false, b, (short)0, acc[g], false, false);
      }
    }

    // Gate nonlinearity + state update, fully in registers (torch i,f,g,o order).
    #pragma unroll
    for (int r = 0; r < 8; ++r) {
      float iv = sigm_f(acc[0][r]);
      float fv = sigm_f(acc[1][r]);
      float gv = tanh_f(acc[2][r]);
      float ov = sigm_f(acc[3][r]);
      float cv = fv * c[r] + iv * gv;
      c[r] = cv;
      float hv = ov * tanh_f(cv);
      pool[r] = fmaxf(pool[r], hv);
      int b = mbase + r + ((lane & 16) ? 8 : 0);
      hwrite[(long)b * HID + hidc] = __float2bfloat16(hv);
    }

    // Grid-wide sense barrier (h writes must be visible to all WGs).
    __threadfence();
    __syncthreads();
    if (threadIdx.x == 0) {
      volatile unsigned* vgen = bar_gen;
      unsigned g0 = *vgen;
      unsigned prev = atomicAdd(bar_cnt, 1u);
      if (prev == gridDim.x - 1) {
        atomicExch(bar_cnt, 0u);
        __threadfence();
        atomicAdd(bar_gen, 1u);
      } else {
        while (*vgen == g0) { __builtin_amdgcn_s_sleep(1); }
      }
      __threadfence();
    }
    __syncthreads();
  }

  #pragma unroll
  for (int r = 0; r < 8; ++r) {
    int b = mbase + r + ((lane & 16) ? 8 : 0);
    hpool[(long)b * HID + hidc] = pool[r];
  }
}

// ---------------------------------------------------------------------------
// Tiny classifier head: out[b][cl] = hpool[b] . W_lin[cl] + b_lin[cl]
// ---------------------------------------------------------------------------
__global__ void head_kernel(const float* __restrict__ hpool, const float* __restrict__ W_lin,
                            const float* __restrict__ b_lin, float* __restrict__ out) {
  int idx = blockIdx.x * blockDim.x + threadIdx.x;
  if (idx >= BATCH * NCLS) return;
  int b = idx / NCLS, cl = idx % NCLS;
  float s = b_lin[cl];
  for (int k = 0; k < HID; ++k) s += hpool[b * HID + k] * W_lin[cl * HID + k];
  out[idx] = s;
}

// ---------------------------------------------------------------------------
extern "C" void kernel_launch(void* const* d_in, const int* in_sizes, int n_in,
                              void* d_out, int out_size, void* d_ws, size_t ws_size,
                              hipStream_t stream) {
  const int*   x      = (const int*)d_in[0];
  const float* embw   = (const float*)d_in[1];
  const float* W_ih   = (const float*)d_in[2];
  const float* W_hh   = (const float*)d_in[3];
  const float* b_ih   = (const float*)d_in[4];
  const float* b_hh   = (const float*)d_in[5];
  const float* W_lin  = (const float*)d_in[6];
  const float* b_lin  = (const float*)d_in[7];
  float* out = (float*)d_out;

  char* ws = (char*)d_ws;
  size_t off = 0;
  auto carve = [&](size_t bytes) -> void* {
    void* p = ws + off;
    off += (bytes + 255) & ~(size_t)255;
    return p;
  };
  bf16_t*  emb16   = (bf16_t*)carve((size_t)MTOT * EPAD * sizeof(bf16_t));   // 20 MB
  bf16_t*  wih16   = (bf16_t*)carve((size_t)G4 * EPAD * sizeof(bf16_t));     // 1.3 MB
  bf16_t*  whh16   = (bf16_t*)carve((size_t)G4 * HID * sizeof(bf16_t));      // 2 MB
  float*   biasSum = (float*)carve((size_t)G4 * sizeof(float));
  float*   xg      = (float*)carve((size_t)SEQ * RWG * XGBLK * sizeof(float)); // 268 MB
  bf16_t*  hbuf    = (bf16_t*)carve(2 * (size_t)BATCH * HID * sizeof(bf16_t));
  float*   hpool   = (float*)carve((size_t)BATCH * HID * sizeof(float));
  unsigned* bar    = (unsigned*)carve(64 * sizeof(unsigned));

  prep_kernel<<<512, 256, 0, stream>>>(W_ih, W_hh, b_ih, b_hh,
                                       wih16, whh16, biasSum, hbuf, bar);
  gather_kernel<<<1024, 256, 0, stream>>>(x, embw, emb16);

  dim3 g1(MTOT / 128, G4 / 64);   // 256 x 32 workgroups
  xproj_kernel<<<g1, 256, 0, stream>>>(emb16, wih16, biasSum, xg);

  lstm_recurrent<<<RWG, 128, 0, stream>>>(whh16, xg, hbuf, hpool, bar, bar + 1);

  head_kernel<<<(BATCH * NCLS + 255) / 256, 256, 0, stream>>>(hpool, W_lin, b_lin, out);
}
